// GeneratorHead_1803886265510
// MI455X (gfx1250) — matrix operand order repaired and checked by reading, not compiled, not run on hardware
//
#include <hip/hip_runtime.h>
#include <hip/hip_bf16.h>
#include <math.h>

// ---------------------------------------------------------------------------
// KAN 2-layer head for MI455X (gfx1250, wave32, WMMA bf16 16x16x32)
//   L1: [4096 x 6912] x [6912 x 128]  (silu + 8 cubic B-spline features)
//   GELU(exact) + LayerNorm(no affine)
//   L2: [4096 x 1152] x [1152 x 30522]
// ---------------------------------------------------------------------------

typedef __attribute__((ext_vector_type(16))) __bf16 v16bf;
typedef __attribute__((ext_vector_type(8)))  __bf16 v8bf;
typedef __attribute__((ext_vector_type(8)))  float  v8f;

#define TOK   4096
#define HDIM  768
#define EDIM  128
#define VDIM  30522
#define NC    9                      // 1 base (silu) + 8 spline coeffs
#define K1    (HDIM*NC)              // 6912
#define K2    (EDIM*NC)              // 1152
#define KT1   (K1/32)                // 216 k-tiles, layer 1
#define KT2   (K2/32)                // 36  k-tiles, layer 2
#define NT1   (EDIM/16)              // 8 n-tiles, layer 1
#define VT2   1920                   // vocab tiles padded: ceil(30522/16)=1908 -> 1920 = 120*16
#define MT    (TOK/16)               // 256 token tiles
#define VBLK  120                    // vocab blocks of 16 tiles (4 tiles / wave)

// workspace layout (bytes): bf16 weight/activation fragment tiles
#define OFF_W1 ((size_t)0)                                   // 8*216*512*2    = 1,769,472
#define OFF_W2 ((size_t)(2*1024*1024))                       // 1920*36*512*2  = 70,778,880
#define OFF_A2 (OFF_W2 + (size_t)VT2*KT2*512*2)              // 256*36*512*2   = 9,437,184
// total ws use ~82.3 MB

// ---------------------------------------------------------------------------
// helpers
// ---------------------------------------------------------------------------
__device__ __forceinline__ unsigned short f2bf(float f) {
  unsigned int u = __float_as_uint(f);
  u += 0x7fffu + ((u >> 16) & 1u);   // round-to-nearest-even
  return (unsigned short)(u >> 16);
}

__device__ __forceinline__ float silu_f(float x) {
  return x / (1.0f + expf(-x));
}

__device__ __forceinline__ float gelu_f(float x) {
  return 0.5f * x * (1.0f + erff(x * 0.70710678118654752f));
}

// Cubic B-spline bases on uniform extended knot grid t_j = -2.2 + 0.4 j (12 knots)
__device__ __forceinline__ void bspline8(float x, float* __restrict__ out) {
  const float h = 0.4f, t0 = -2.2f;
  float b[11];
#pragma unroll
  for (int j = 0; j < 11; ++j) {
    float tj  = t0 + h * (float)j;
    float tj1 = tj + h;
    b[j] = (x >= tj && x < tj1) ? 1.0f : 0.0f;
  }
#pragma unroll
  for (int p = 1; p <= 3; ++p) {
    float inv = 1.0f / (h * (float)p);
#pragma unroll
    for (int j = 0; j < 11 - 1; ++j) {
      if (j < 11 - p) {
        float tj   = t0 + h * (float)j;
        float tjp1 = t0 + h * (float)(j + p + 1);
        b[j] = (x - tj) * inv * b[j] + (tjp1 - x) * inv * b[j + 1];
      }
    }
  }
#pragma unroll
  for (int j = 0; j < 8; ++j) out[j] = b[j];
}

// A-matrix fragment swizzle: 16-bit A 16x32, lanes 0-15 hold K{0..7,16..23},
// lanes 16-31 hold K{8..15,24..31}; returns (lane*16 + j) for (row, k_local)
__device__ __forceinline__ int a_tile_off(int row, int k_local) {
  int hi   = (k_local >> 3) & 1;                 // 8..15 / 24..31 -> hi lanes
  int lane = row + (hi ? 16 : 0);
  int j    = (k_local & 7) + ((k_local >= 16) ? 8 : 0);
  return lane * 16 + j;
}

// ---------------------------------------------------------------------------
// Weight tiling: fp32 (base_w [cols,in], spline_w [cols,in,8]) -> bf16 tiles in
// B-matrix fragment layout: tile(nt,kt): elem[lane*16+j] = W[nt*16+lane%16]
//                                   [kt*32+(lane/16)*8+(j%8)+(j/8)*16]
// augmented k = e*9 + c  (c==0 -> base weight, c>0 -> spline coeff c-1)
// ---------------------------------------------------------------------------
__global__ __launch_bounds__(256) void build_w_tiles(
    const float* __restrict__ base_w, const float* __restrict__ spline_w,
    unsigned short* __restrict__ wt, int in_dim, int n_cols, int n_kt) {
  int tile = blockIdx.x;
  int kt = tile % n_kt;
  int nt = tile / n_kt;
  for (int idx = threadIdx.x; idx < 512; idx += blockDim.x) {
    int lane = idx >> 4, j = idx & 15;
    int col = nt * 16 + (lane & 15);
    int k   = kt * 32 + ((lane >> 4) << 3) + (j & 7) + ((j >> 3) << 4);
    int e   = k / NC, c = k - e * NC;
    float v = 0.0f;
    if (col < n_cols) {
      v = (c == 0) ? base_w[(size_t)col * in_dim + e]
                   : spline_w[((size_t)col * in_dim + e) * 8 + (c - 1)];
    }
    wt[(size_t)tile * 512 + idx] = f2bf(v);
  }
}

// ---------------------------------------------------------------------------
// Layer 1: features->LDS, bf16 WMMA over K=6912, fused GELU + LayerNorm,
// then emit layer-2 bf16 A-fragments (silu + splines of normalized h) to ws.
// Block: 128 threads (4 waves), one 16-token tile, all 128 outputs.
// ---------------------------------------------------------------------------
__global__ __launch_bounds__(128) void kan_layer1(
    const float* __restrict__ hidden,          // [4096,768] fp32
    const unsigned short* __restrict__ w1t,    // tiled bf16 weights
    unsigned short* __restrict__ a2t) {        // out: tiled bf16 A2 fragments
  __shared__ unsigned short feat[16][288];     // 32 dims * 9 features / chunk
  __shared__ float htile[16][EDIM];            // gelu(h) staging for LN
  __shared__ float red[16][8][2];              // LN partial sums
  __shared__ float mu_s[16], rs_s[16];

  const int mt   = blockIdx.x;                 // token tile
  const int tid  = threadIdx.x;
  const int lane = tid & 31;
  const int wave = tid >> 5;
  const int nt0  = wave * 2, nt1 = wave * 2 + 1;

  v8f acc0 = {}, acc1 = {};

  const int tok_l = tid >> 3;                  // 0..15
  const int dpack = (tid & 7) * 4;             // 4 dims per thread
  const int row   = lane & 15;
  const int klo   = (lane >> 4) * 8;

  for (int ch = 0; ch < HDIM / 32; ++ch) {     // 24 chunks of 32 input dims
    // ---- produce features for this chunk into LDS ----
    const float* xp = hidden + ((size_t)(mt * 16 + tok_l)) * HDIM + ch * 32 + dpack;
    float4 x4 = *(const float4*)xp;
    float xs[4] = {x4.x, x4.y, x4.z, x4.w};
#pragma unroll
    for (int q = 0; q < 4; ++q) {
      float x = xs[q];
      int kl = (dpack + q) * NC;
      feat[tok_l][kl] = f2bf(silu_f(x));
      float bs[8];
      bspline8(x, bs);
#pragma unroll
      for (int c = 0; c < 8; ++c) feat[tok_l][kl + 1 + c] = f2bf(bs[c]);
    }
    __syncthreads();

    // ---- 9 WMMA k-steps over the 288 chunk features ----
#pragma unroll
    for (int kk = 0; kk < 9; ++kk) {
      int kt = ch * 9 + kk;
      v8bf alo = *(const v8bf*)&feat[row][kk * 32 + klo];
      v8bf ahi = *(const v8bf*)&feat[row][kk * 32 + 16 + klo];
      v16bf a;
#pragma unroll
      for (int i = 0; i < 8; ++i) { a[i] = alo[i]; a[8 + i] = ahi[i]; }
      v16bf b0 = *(const v16bf*)(w1t + ((size_t)(nt0 * KT1 + kt)) * 512 + lane * 16);
      v16bf b1 = *(const v16bf*)(w1t + ((size_t)(nt1 * KT1 + kt)) * 512 + lane * 16);
      acc0 = __builtin_amdgcn_wmma_f32_16x16x32_bf16(false, a, false, b0,
                                                     (short)0, acc0, false, false);
      acc1 = __builtin_amdgcn_wmma_f32_16x16x32_bf16(false, a, false, b1,
                                                     (short)0, acc1, false, false);
    }
    __syncthreads();                           // protect feat for next chunk
  }

  // ---- exact GELU, stage to LDS for LayerNorm ----
#pragma unroll
  for (int r = 0; r < 8; ++r) {
    int m = r + (lane >> 4) * 8;
    htile[m][nt0 * 16 + (lane & 15)] = gelu_f(acc0[r]);
    htile[m][nt1 * 16 + (lane & 15)] = gelu_f(acc1[r]);
  }
  __syncthreads();

  // ---- LayerNorm (no affine), EPS = 1e-12 ----
  const int token = tid >> 3;                  // 0..15
  const int seg   = tid & 7;                   // 16 columns each
  float sum = 0.0f, sq = 0.0f;
#pragma unroll
  for (int c = 0; c < 16; ++c) {
    float v = htile[token][seg * 16 + c];
    sum += v; sq += v * v;
  }
  red[token][seg][0] = sum;
  red[token][seg][1] = sq;
  __syncthreads();
  if (tid < 16) {
    float s = 0.0f, s2 = 0.0f;
#pragma unroll
    for (int g = 0; g < 8; ++g) { s += red[tid][g][0]; s2 += red[tid][g][1]; }
    float mu  = s * (1.0f / EDIM);
    float var = s2 * (1.0f / EDIM) - mu * mu;
    mu_s[tid] = mu;
    rs_s[tid] = rsqrtf(var + 1e-12f);
  }
  __syncthreads();

  // ---- build layer-2 A fragments (silu + splines of normalized h) ----
  const float mu = mu_s[token], rs = rs_s[token];
  unsigned short* a2base = a2t + ((size_t)mt * KT2) * 512;
#pragma unroll 4
  for (int c = 0; c < 16; ++c) {
    int e = seg * 16 + c;
    float x = (htile[token][e] - mu) * rs;
    float fv[9];
    fv[0] = silu_f(x);
    bspline8(x, &fv[1]);
#pragma unroll
    for (int cc = 0; cc < 9; ++cc) {
      int k  = e * NC + cc;
      int kt = k >> 5, kl = k & 31;
      a2base[(size_t)kt * 512 + a_tile_off(token, kl)] = f2bf(fv[cc]);
    }
  }
}

// ---------------------------------------------------------------------------
// Layer 2: pure bf16 WMMA GEMM [4096 x 1152] x [1152 x 30522(+pad)]
// Block: 128 threads / 4 waves; 4 vocab n-tiles per wave (16 per block):
// each A-fragment load feeds 4 WMMAs (inner loop = 5 loads : 4 WMMAs).
// ---------------------------------------------------------------------------
__global__ __launch_bounds__(128) void kan_layer2(
    const unsigned short* __restrict__ a2t,
    const unsigned short* __restrict__ w2t,
    float* __restrict__ out) {
  const int mt   = blockIdx.x;                 // 0..255
  const int blk  = blockIdx.y;                 // 0..119
  const int tid  = threadIdx.x;
  const int lane = tid & 31;
  const int wave = tid >> 5;
  const int vt0  = blk * 16 + wave * 4;        // 4 consecutive tiles, < 1920

  v8f acc0 = {}, acc1 = {}, acc2 = {}, acc3 = {};
  const unsigned short* ap  = a2t + ((size_t)mt * KT2) * 512 + lane * 16;
  const unsigned short* bp0 = w2t + ((size_t)(vt0 + 0) * KT2) * 512 + lane * 16;
  const unsigned short* bp1 = w2t + ((size_t)(vt0 + 1) * KT2) * 512 + lane * 16;
  const unsigned short* bp2 = w2t + ((size_t)(vt0 + 2) * KT2) * 512 + lane * 16;
  const unsigned short* bp3 = w2t + ((size_t)(vt0 + 3) * KT2) * 512 + lane * 16;

  for (int kt = 0; kt < KT2; ++kt) {
    v16bf a  = *(const v16bf*)(ap  + (size_t)kt * 512);
    v16bf b0 = *(const v16bf*)(bp0 + (size_t)kt * 512);
    v16bf b1 = *(const v16bf*)(bp1 + (size_t)kt * 512);
    v16bf b2 = *(const v16bf*)(bp2 + (size_t)kt * 512);
    v16bf b3 = *(const v16bf*)(bp3 + (size_t)kt * 512);
    if (kt + 1 < KT2) {
      __builtin_prefetch(bp0 + (size_t)(kt + 1) * 512, 0, 1);
      __builtin_prefetch(bp2 + (size_t)(kt + 1) * 512, 0, 1);
    }
    acc0 = __builtin_amdgcn_wmma_f32_16x16x32_bf16(false, a, false, b0,
                                                   (short)0, acc0, false, false);
    acc1 = __builtin_amdgcn_wmma_f32_16x16x32_bf16(false, a, false, b1,
                                                   (short)0, acc1, false, false);
    acc2 = __builtin_amdgcn_wmma_f32_16x16x32_bf16(false, a, false, b2,
                                                   (short)0, acc2, false, false);
    acc3 = __builtin_amdgcn_wmma_f32_16x16x32_bf16(false, a, false, b3,
                                                   (short)0, acc3, false, false);
  }

  const int colb = lane & 15;
  const int rowo = (lane >> 4) * 8;
#pragma unroll
  for (int r = 0; r < 8; ++r) {
    int token = mt * 16 + r + rowo;
    float* orow = out + (size_t)token * VDIM;
    int c0 = (vt0 + 0) * 16 + colb;
    int c1 = (vt0 + 1) * 16 + colb;
    int c2 = (vt0 + 2) * 16 + colb;
    int c3 = (vt0 + 3) * 16 + colb;
    if (c0 < VDIM) orow[c0] = acc0[r];
    if (c1 < VDIM) orow[c1] = acc1[r];
    if (c2 < VDIM) orow[c2] = acc2[r];
    if (c3 < VDIM) orow[c3] = acc3[r];
  }
}

// ---------------------------------------------------------------------------
extern "C" void kernel_launch(void* const* d_in, const int* in_sizes, int n_in,
                              void* d_out, int out_size, void* d_ws, size_t ws_size,
                              hipStream_t stream) {
  const float* hidden    = (const float*)d_in[0];   // [8,512,768]
  const float* base_w1   = (const float*)d_in[1];   // [128,768]
  const float* spline_w1 = (const float*)d_in[2];   // [128,768,8]
  const float* base_w2   = (const float*)d_in[3];   // [30522,128]
  const float* spline_w2 = (const float*)d_in[4];   // [30522,128,8]
  float* out = (float*)d_out;

  char* ws = (char*)d_ws;
  unsigned short* w1t = (unsigned short*)(ws + OFF_W1);
  unsigned short* w2t = (unsigned short*)(ws + OFF_W2);
  unsigned short* a2t = (unsigned short*)(ws + OFF_A2);

  // 1) tile weights to bf16 B-fragments (w2 tiles stay resident in 192MB L2)
  build_w_tiles<<<NT1 * KT1, 256, 0, stream>>>(base_w1, spline_w1, w1t,
                                               HDIM, EDIM, KT1);
  build_w_tiles<<<VT2 * KT2, 256, 0, stream>>>(base_w2, spline_w2, w2t,
                                               EDIM, VDIM, KT2);
  // 2) layer 1 + GELU + LN + layer-2 feature fragments
  kan_layer1<<<MT, 128, 0, stream>>>(hidden, w1t, a2t);
  // 3) layer 2 GEMM -> logits
  kan_layer2<<<dim3(MT, VBLK), 128, 0, stream>>>(a2t, w2t, out);
}
